// HGTLayer_850403524852
// MI455X (gfx1250) — matrix operand order repaired and checked by reading, not compiled, not run on hardware
//
#include <hip/hip_runtime.h>

// ---------------------------------------------------------------------------
// HGT layer for MI455X (gfx1250, wave32, WMMA).
// GEMMs: bf16x3 split-precision WMMA (v_wmma_f32_16x16x32_bf16) with
// pre-split bf16 hi/lo operands. B is pre-split TRANSPOSED (column-major,
// K-contiguous) so the per-K-chunk LDS stage is a pure wide copy done with
// CDNA5 async global->LDS loads (global_load_async_to_lds_b128, ASYNCcnt).
//   A = Ahi + Alo, B = Bhi + Blo  ->  C += Alo*Bhi + Ahi*Blo + Ahi*Bhi
// ---------------------------------------------------------------------------

#define HN    20000      // nodes per type
#define HH    256        // hidden
#define HEADS 8
#define HD    32
#define HE    320000     // edges per relation
#define HFF   1024

typedef __attribute__((ext_vector_type(16))) __bf16         v16bf;
typedef __attribute__((ext_vector_type(8)))  float          v8f;
typedef __attribute__((ext_vector_type(8)))  unsigned short ush8;
typedef __attribute__((ext_vector_type(16))) unsigned short ush16;

// ---------------------------------------------------------------------------
// Split fp32 -> bf16 hi/lo pair (round-to-nearest-even):  x ~= hi + lo
// ---------------------------------------------------------------------------
__global__ __launch_bounds__(256) void split_bf16_kernel(
    const float* __restrict__ src, unsigned short* __restrict__ hi,
    unsigned short* __restrict__ lo, long long n)
{
    long long i = (long long)blockIdx.x * blockDim.x + threadIdx.x;
    if (i >= n) return;
    float v = src[i];
    __bf16 h = (__bf16)v;
    __bf16 l = (__bf16)(v - (float)h);
    hi[i] = __builtin_bit_cast(unsigned short, h);
    lo[i] = __builtin_bit_cast(unsigned short, l);
}

// ---------------------------------------------------------------------------
// Split fp32 [K x N] row-major -> bf16 hi/lo TRANSPOSED [N x K] (K-contig)
// ---------------------------------------------------------------------------
__global__ __launch_bounds__(256) void split_bf16_t_kernel(
    const float* __restrict__ src, unsigned short* __restrict__ hi,
    unsigned short* __restrict__ lo, int K, int N)
{
    long long i = (long long)blockIdx.x * blockDim.x + threadIdx.x;
    if (i >= (long long)K * N) return;
    int k = (int)(i / N);
    int n = (int)(i - (long long)k * N);
    float v = src[i];
    __bf16 h = (__bf16)v;
    __bf16 l = (__bf16)(v - (float)h);
    size_t o = (size_t)n * K + k;
    hi[o] = __builtin_bit_cast(unsigned short, h);
    lo[o] = __builtin_bit_cast(unsigned short, l);
}

// ---------------------------------------------------------------------------
// Fold per-head relation transform into projection weight + bias.
// Weff[t][r][h*32+o] = sum_i W[t][r][h*32+i] * R[t][h][i][o]   (r==256 -> bias)
// ---------------------------------------------------------------------------
__global__ __launch_bounds__(256) void fold_rel_kernel(
    const float* __restrict__ W, const float* __restrict__ b,
    const float* __restrict__ R, float* __restrict__ Weff,
    float* __restrict__ beff)
{
    int idx = blockIdx.x * blockDim.x + threadIdx.x;
    const int per_t = 257 * HH;
    if (idx >= 2 * per_t) return;
    int t = idx / per_t;
    int rem = idx - t * per_t;
    int r = rem / HH;           // 0..256 (256 == bias row)
    int c = rem - r * HH;
    int h = c >> 5, o = c & 31;
    const float* src = (r < HH) ? (W + (size_t)t * HH * HH + (size_t)r * HH)
                                : (b + (size_t)t * HH);
    const float* Rp = R + ((size_t)t * HEADS + h) * HD * HD;  // [i][o]
    float acc = 0.f;
#pragma unroll
    for (int i = 0; i < HD; ++i)
        acc += src[h * HD + i] * Rp[i * HD + o];
    if (r < HH) Weff[(size_t)t * HH * HH + (size_t)r * HH + c] = acc;
    else        beff[(size_t)t * HH + c] = acc;
}

__global__ __launch_bounds__(256) void fill_zero_kernel(float* __restrict__ p, long long n)
{
    long long i = (long long)blockIdx.x * blockDim.x + threadIdx.x;
    if (i < n) p[i] = 0.f;
}

// ---------------------------------------------------------------------------
// bf16x3 WMMA GEMM:  C[M x N] = (Ahi+Alo)[M x K] @ (Bhi+Blo) + bias[N]
// A row-major bf16 pair; B TRANSPOSED bf16 pair [N x K] (K-contiguous).
// Workgroup: 8 waves, 32x256 output tile (2 waves M x 4 waves N; each wave
// 16 rows x 64 cols = 4 accumulators). K stepped by 32. Per step, the
// 32x256 B chunk (already K-contiguous per column) is staged into LDS with
// async global->LDS b128 copies, then each lane reads its 16-element B
// fragment as one contiguous 32B ds_load pair.
// Requires M%32==0, N%256==0, K%32==0 (all shapes here divide exactly, so
// EXEC is always all-ones for WMMA).
// Register layouts (A/B: contiguous K order used here, not the ISA table; C per CDNA5 ISA 7.12.2):
//   A 16x32 bf16 : lanes 0-15 row M=lane, elems 0..15 -> K 0..15 (contiguous)
//                  lanes 16-31 row M=lane-16, elems 0..15 -> K 16..31
//   B 32x16 bf16 : lane = col (mod 16); elems 0..15 -> ascending K within half
//   C 16x16 f32  : vgpr r, lane l: row = r + 8*(l>=16), col = l&15
// ---------------------------------------------------------------------------
__global__ __launch_bounds__(256) void gemm_bf16x3_kernel(
    const unsigned short* __restrict__ Ahi, const unsigned short* __restrict__ Alo,
    const unsigned short* __restrict__ Bthi, const unsigned short* __restrict__ Btlo,
    const float* __restrict__ bias, float* __restrict__ C,
    int M, int N, int K)
{
    __shared__ unsigned short ldsBhi[32 * 256];   // [col][k] col-major, 16KB
    __shared__ unsigned short ldsBlo[32 * 256];   // 16KB

    int tilesN = N >> 8;
    int bm = blockIdx.x / tilesN;
    int bn = blockIdx.x - bm * tilesN;
    int tm = bm << 5;            // 32-row block tile
    int tn = bn << 8;            // 256-col block tile

    int tid   = threadIdx.x;
    int wave  = tid >> 5;        // 0..7
    int lane  = tid & 31;
    int wm    = wave >> 2;       // 0..1  (M sub-tile)
    int wn    = wave & 3;        // 0..3  (N sub-tile, 64 cols each)
    int l15   = lane & 15;
    int khalf = lane >> 4;

    int rowA  = tm + (wm << 4) + l15;          // this lane's A row
    int cbase = tn + (wn << 6);                // wave's first column

    // accumulators for 4 16x16 C tiles, bias-initialized
    v8f acc[4];
#pragma unroll
    for (int nt = 0; nt < 4; ++nt) {
        float bv = bias ? bias[cbase + (nt << 4) + l15] : 0.f;
#pragma unroll
        for (int r = 0; r < 8; ++r) acc[nt][r] = bv;
    }

    const unsigned short* Arow_hi = Ahi + (size_t)rowA * K;
    const unsigned short* Arow_lo = Alo + (size_t)rowA * K;
    // this thread's staging column (one column's 32-K chunk per component)
    const unsigned short* colH = Bthi + (size_t)(tn + tid) * K;
    const unsigned short* colL = Btlo + (size_t)(tn + tid) * K;
    unsigned ldsH = (unsigned)(size_t)&ldsBhi[tid << 5];   // 64B per column
    unsigned ldsL = (unsigned)(size_t)&ldsBlo[tid << 5];

    for (int k0 = 0; k0 < K; k0 += 32) {
        // ---- async stage of B chunk [256 cols x 32 K] into LDS ----
        __syncthreads();   // previous iteration's LDS reads complete
#pragma unroll
        for (int j = 0; j < 4; ++j) {
            asm volatile("global_load_async_to_lds_b128 %0, %1, off"
                         :: "v"(ldsH + j * 16),
                            "v"((unsigned long long)(size_t)(colH + k0 + j * 8))
                         : "memory");
            asm volatile("global_load_async_to_lds_b128 %0, %1, off"
                         :: "v"(ldsL + j * 16),
                            "v"((unsigned long long)(size_t)(colL + k0 + j * 8))
                         : "memory");
        }
        asm volatile("s_wait_asynccnt 0" ::: "memory");
        __syncthreads();

        // ---- A fragments (two aligned 16B chunks per component) ----
        const unsigned short* aph = Arow_hi + k0 + khalf * 16;
        const unsigned short* apl = Arow_lo + k0 + khalf * 16;
        ush8 h0 = *(const ush8*)(aph);
        ush8 h1 = *(const ush8*)(aph + 8);
        ush8 l0 = *(const ush8*)(apl);
        ush8 l1 = *(const ush8*)(apl + 8);
        ush16 ua, ul;
#pragma unroll
        for (int j = 0; j < 8; ++j) {
            ua[j] = h0[j]; ua[j + 8] = h1[j];
            ul[j] = l0[j]; ul[j + 8] = l1[j];
        }
        v16bf a_hi = __builtin_bit_cast(v16bf, ua);
        v16bf a_lo = __builtin_bit_cast(v16bf, ul);

        // ---- 4 N-subtiles: one contiguous 32B LDS read per B component ----
#pragma unroll
        for (int nt = 0; nt < 4; ++nt) {
            int lc = (wn << 6) + (nt << 4) + l15;          // column within tile
            const ush16* bph = (const ush16*)&ldsBhi[(lc << 5) + khalf * 16];
            const ush16* bpl = (const ush16*)&ldsBlo[(lc << 5) + khalf * 16];
            v16bf b_hi = __builtin_bit_cast(v16bf, *bph);
            v16bf b_lo = __builtin_bit_cast(v16bf, *bpl);

            acc[nt] = __builtin_amdgcn_wmma_f32_16x16x32_bf16(
                false, a_lo, false, b_hi, (short)0, acc[nt], false, false);
            acc[nt] = __builtin_amdgcn_wmma_f32_16x16x32_bf16(
                false, a_hi, false, b_lo, (short)0, acc[nt], false, false);
            acc[nt] = __builtin_amdgcn_wmma_f32_16x16x32_bf16(
                false, a_hi, false, b_hi, (short)0, acc[nt], false, false);
        }
    }

    // ---- store C ----
#pragma unroll
    for (int nt = 0; nt < 4; ++nt) {
        float* cp = C + (size_t)(tm + (wm << 4) + khalf * 8) * N
                      + cbase + (nt << 4) + l15;
#pragma unroll
        for (int r = 0; r < 8; ++r) cp[(size_t)r * N] = acc[nt][r];
    }
}

// ---------------------------------------------------------------------------
// Edge attention logits + segment-max (int-pattern atomicMax; buffer init 0
// matches the reference's max(.,0) clamp and is exact for the >=0 lattice).
// ---------------------------------------------------------------------------
__global__ __launch_bounds__(256) void edge_att_kernel(
    const float* __restrict__ kp, const float* __restrict__ q,
    const int* __restrict__ ei, const float* __restrict__ pri,
    float* __restrict__ att, int* __restrict__ amax)
{
    long long t = (long long)blockIdx.x * blockDim.x + threadIdx.x;
    if (t >= (long long)HE * HEADS) return;
    int e = (int)(t >> 3);
    int h = (int)(t & 7);
    int s = ei[e];
    int d = ei[HE + e];
    const float* kr = kp + (size_t)s * HH + h * HD;
    const float* qr = q  + (size_t)d * HH + h * HD;
    float acc = 0.f;
#pragma unroll
    for (int i = 0; i < HD; ++i) acc += kr[i] * qr[i];
    float a = acc * 0.17677669529663689f /* 1/sqrt(32) */ + pri[h];
    att[(size_t)e * HEADS + h] = a;
    atomicMax(&amax[d * HEADS + h], __float_as_int(a));
}

// ---------------------------------------------------------------------------
// Edge scatter: p = exp(att - amax[d]); sum[d,h] += p; macc[d,:] += p*msg[s,:]
// (unnormalized; normalized afterwards). One thread per (edge, channel).
// ---------------------------------------------------------------------------
__global__ __launch_bounds__(256) void edge_scatter_kernel(
    const float* __restrict__ msgp, const float* __restrict__ att,
    const int* __restrict__ ei, const int* __restrict__ amax,
    float* __restrict__ sum, float* __restrict__ macc)
{
    long long t = (long long)blockIdx.x * blockDim.x + threadIdx.x;
    if (t >= (long long)HE * HH) return;
    int e = (int)(t >> 8);
    int c = (int)(t & 255);
    int h = c >> 5;
    int s = ei[e];
    int d = ei[HE + e];
    float a  = att[(size_t)e * HEADS + h];
    float mx = __int_as_float(amax[d * HEADS + h]);
    float p  = expf(a - mx);
    if ((c & 31) == 0) atomicAdd(&sum[d * HEADS + h], p);
    atomicAdd(&macc[(size_t)d * HH + c], p * msgp[(size_t)s * HH + c]);
}

// ---------------------------------------------------------------------------
// macc /= max(sum, 1e-8)  (both node types, in place)
// ---------------------------------------------------------------------------
__global__ __launch_bounds__(256) void normalize_kernel(
    float* __restrict__ macc, const float* __restrict__ sum)
{
    long long t = (long long)blockIdx.x * blockDim.x + threadIdx.x;
    if (t >= (long long)2 * HN * HH) return;
    int n = (int)(t >> 8);
    int c = (int)(t & 255);
    int h = c >> 5;
    macc[t] /= fmaxf(sum[n * HEADS + h], 1e-8f);
}

// ---------------------------------------------------------------------------
// y = LayerNorm(x + o) * g + b     (one wave per row of 256)
// ---------------------------------------------------------------------------
__global__ __launch_bounds__(256) void add_ln_kernel(
    const float* __restrict__ x, const float* __restrict__ o,
    const float* __restrict__ g, const float* __restrict__ b,
    float* __restrict__ y, int nrows)
{
    int row  = (blockIdx.x * blockDim.x + threadIdx.x) >> 5;
    int lane = threadIdx.x & 31;
    if (row >= nrows) return;
    const float* xr = x + (size_t)row * HH;
    const float* orr = o + (size_t)row * HH;
    float v[8];
    float s = 0.f;
#pragma unroll
    for (int j = 0; j < 8; ++j) {
        v[j] = xr[lane + 32 * j] + orr[lane + 32 * j];
        s += v[j];
    }
#pragma unroll
    for (int off = 16; off; off >>= 1) s += __shfl_xor(s, off, 32);
    float mu = s * (1.f / HH);
    float var = 0.f;
#pragma unroll
    for (int j = 0; j < 8; ++j) { float d = v[j] - mu; var += d * d; }
#pragma unroll
    for (int off = 16; off; off >>= 1) var += __shfl_xor(var, off, 32);
    float rs = rsqrtf(var * (1.f / HH) + 1e-5f);
#pragma unroll
    for (int j = 0; j < 8; ++j) {
        int c = lane + 32 * j;
        y[(size_t)row * HH + c] = (v[j] - mu) * rs * g[c] + b[c];
    }
}

// ---------------------------------------------------------------------------
// Exact GELU (erf form) in place
// ---------------------------------------------------------------------------
__global__ __launch_bounds__(256) void gelu_kernel(float* __restrict__ p, long long n)
{
    long long i = (long long)blockIdx.x * blockDim.x + threadIdx.x;
    if (i < n) {
        float x = p[i];
        p[i] = 0.5f * x * (1.f + erff(x * 0.70710678118654752f));
    }
}

// ---------------------------------------------------------------------------
// Host-side launcher
// ---------------------------------------------------------------------------
static inline int cdiv(long long a, int b) { return (int)((a + b - 1) / b); }

extern "C" void kernel_launch(void* const* d_in, const int* in_sizes, int n_in,
                              void* d_out, int out_size, void* d_ws, size_t ws_size,
                              hipStream_t stream)
{
    (void)in_sizes; (void)n_in; (void)out_size; (void)ws_size;

    const float* x_a     = (const float*)d_in[0];
    const float* x_b     = (const float*)d_in[1];
    const float* Wk      = (const float*)d_in[2];
    const float* bk      = (const float*)d_in[3];
    const float* Wq      = (const float*)d_in[4];
    const float* bq      = (const float*)d_in[5];
    const float* Wv      = (const float*)d_in[6];
    const float* bv      = (const float*)d_in[7];
    const float* rel_att = (const float*)d_in[8];
    const float* rel_msg = (const float*)d_in[9];
    const float* rel_pri = (const float*)d_in[10];
    const float* Wo      = (const float*)d_in[11];
    const float* bo      = (const float*)d_in[12];
    const float* ln1_g   = (const float*)d_in[13];
    const float* ln1_b   = (const float*)d_in[14];
    const float* W1      = (const float*)d_in[15];
    const float* b1      = (const float*)d_in[16];
    const float* W2      = (const float*)d_in[17];
    const float* b2      = (const float*)d_in[18];
    const float* ln2_g   = (const float*)d_in[19];
    const float* ln2_b   = (const float*)d_in[20];
    const int*   ei_ab   = (const int*)d_in[21];
    const int*   ei_ba   = (const int*)d_in[22];
    float* out = (float*)d_out;

    // ---- workspace layout ----
    const size_t NH = (size_t)HN * HH;         // 5,120,000
    const size_t E8 = (size_t)HE * HEADS;      // 2,560,000
    float* ws    = (float*)d_ws;
    float* wkeff = ws;                          // 2*256*256
    float* bkeff = wkeff + 2 * HH * HH;         // 2*256
    float* wveff = bkeff + 2 * HH;              // 2*256*256
    float* bveff = wveff + 2 * HH * HH;         // 2*256
    float* kp    = bveff + 2 * HH;              // 2*N*H  (K'; later o)
    float* vp    = kp  + 2 * NH;                // 2*N*H  (V'; later h2)
    float* qb    = vp  + 2 * NH;                // 2*N*H  (Q;  later x1)
    float* attb  = qb  + 2 * NH;                // 2*E*8
    int*   amax  = (int*)(attb + 2 * E8);       // 2*N*8
    float* sums  = (float*)amax + 2 * (size_t)HN * HEADS;  // 2*N*8
    float* macc  = sums + 2 * (size_t)HN * HEADS;          // 2*N*H
    float* ffnb  = macc + 2 * NH;               // N*FF (per-type, reused)
    // bf16 split regions (ushort)
    unsigned short* u = (unsigned short*)(ffnb + (size_t)HN * HFF);
    unsigned short* xs_hi = u;                  // 2*N*H (x_a then x_b)
    unsigned short* xs_lo = xs_hi + 2 * NH;     // 2*N*H
    unsigned short* as_hi = xs_lo + 2 * NH;     // max M*K = 20000*1024
    unsigned short* as_lo = as_hi + (size_t)HN * HFF;
    unsigned short* bs_hi = as_lo + (size_t)HN * HFF;      // max K*N = 256*1024
    unsigned short* bs_lo = bs_hi + (size_t)HH * HFF;

    const long long szHH  = (long long)HH * HH;    // 65536
    const long long szHF  = (long long)HH * HFF;   // 262144

    // 1) fold relation transforms into K/V projections
    {
        int tot = 2 * 257 * HH;
        fold_rel_kernel<<<cdiv(tot, 256), 256, 0, stream>>>(Wk, bk, rel_att, wkeff, bkeff);
        fold_rel_kernel<<<cdiv(tot, 256), 256, 0, stream>>>(Wv, bv, rel_msg, wveff, bveff);
    }

    // 2) zero accumulators (amax, sums, macc are contiguous)
    {
        long long nz = 2LL * HN * HEADS * 2 + 2LL * NH;
        fill_zero_kernel<<<cdiv(nz, 256), 256, 0, stream>>>((float*)amax, nz);
    }

    // 3) pre-split node features (row-major A operands)
    split_bf16_kernel<<<cdiv(NH, 256), 256, 0, stream>>>(x_a, xs_hi, xs_lo, NH);
    split_bf16_kernel<<<cdiv(NH, 256), 256, 0, stream>>>(x_b, xs_hi + NH, xs_lo + NH, NH);

    // 4) projections: K' = x@Wkeff+bkeff, V' = x@Wveff+bveff, Q = x@Wq+bq
    {
        int blocksW = cdiv(szHH, 256);
        int gemmGrid = (HN / 32) * (HH / 256);   // 625
        for (int t = 0; t < 2; ++t) {
            const unsigned short* xh = xs_hi + (size_t)t * NH;
            const unsigned short* xl = xs_lo + (size_t)t * NH;
            split_bf16_t_kernel<<<blocksW, 256, 0, stream>>>(
                wkeff + t * szHH, bs_hi, bs_lo, HH, HH);
            gemm_bf16x3_kernel<<<gemmGrid, 256, 0, stream>>>(
                xh, xl, bs_hi, bs_lo, bkeff + (size_t)t * HH,
                kp + (size_t)t * NH, HN, HH, HH);
            split_bf16_t_kernel<<<blocksW, 256, 0, stream>>>(
                wveff + t * szHH, bs_hi, bs_lo, HH, HH);
            gemm_bf16x3_kernel<<<gemmGrid, 256, 0, stream>>>(
                xh, xl, bs_hi, bs_lo, bveff + (size_t)t * HH,
                vp + (size_t)t * NH, HN, HH, HH);
            split_bf16_t_kernel<<<blocksW, 256, 0, stream>>>(
                Wq + t * szHH, bs_hi, bs_lo, HH, HH);
            gemm_bf16x3_kernel<<<gemmGrid, 256, 0, stream>>>(
                xh, xl, bs_hi, bs_lo, bq + (size_t)t * HH,
                qb + (size_t)t * NH, HN, HH, HH);
        }
    }

    // 5) edge attention logits + segment max
    {
        int blocks = cdiv((long long)HE * HEADS, 256);
        // relation 0: a -> b   (src K'_a, dst Q_b, dst accum type b)
        edge_att_kernel<<<blocks, 256, 0, stream>>>(
            kp, qb + NH, ei_ab, rel_pri, attb, amax + (size_t)HN * HEADS);
        // relation 1: b -> a
        edge_att_kernel<<<blocks, 256, 0, stream>>>(
            kp + NH, qb, ei_ba, rel_pri + HEADS, attb + E8, amax);
    }

    // 6) edge scatter: exp, denom sum, weighted message accumulate
    {
        int blocks = cdiv((long long)HE * HH, 256);
        edge_scatter_kernel<<<blocks, 256, 0, stream>>>(
            vp, attb, ei_ab, amax + (size_t)HN * HEADS,
            sums + (size_t)HN * HEADS, macc + NH);
        edge_scatter_kernel<<<blocks, 256, 0, stream>>>(
            vp + NH, attb + E8, ei_ba, amax, sums, macc);
    }

    // 7) normalize messages
    normalize_kernel<<<cdiv(2LL * NH, 256), 256, 0, stream>>>(macc, sums);

    // 8) node update per type
    {
        int gridH = (HN / 32) * (HH / 256);      // 625
        int gridF = (HN / 32) * (HFF / 256);     // 2500
        int lnBlocks = cdiv((long long)HN * 32, 256);

        for (int t = 0; t < 2; ++t) {
            const float* xt = t ? x_b : x_a;
            float* o_t  = kp + (size_t)t * NH;   // reuse K' space
            float* x1_t = qb + (size_t)t * NH;   // reuse Q space
            float* h2_t = vp + (size_t)t * NH;   // reuse V' space

            // o = m @ Wo + bo
            split_bf16_kernel<<<cdiv(NH, 256), 256, 0, stream>>>(
                macc + (size_t)t * NH, as_hi, as_lo, NH);
            split_bf16_t_kernel<<<cdiv(szHH, 256), 256, 0, stream>>>(
                Wo + t * szHH, bs_hi, bs_lo, HH, HH);
            gemm_bf16x3_kernel<<<gridH, 256, 0, stream>>>(
                as_hi, as_lo, bs_hi, bs_lo, bo + (size_t)t * HH, o_t, HN, HH, HH);

            // x1 = LN1(x + o)
            add_ln_kernel<<<lnBlocks, 256, 0, stream>>>(
                xt, o_t, ln1_g + (size_t)t * HH, ln1_b + (size_t)t * HH, x1_t, HN);

            // f = gelu(x1 @ W1 + b1)
            split_bf16_kernel<<<cdiv(NH, 256), 256, 0, stream>>>(
                x1_t, as_hi, as_lo, NH);
            split_bf16_t_kernel<<<cdiv(szHF, 256), 256, 0, stream>>>(
                W1 + t * szHF, bs_hi, bs_lo, HH, HFF);
            gemm_bf16x3_kernel<<<gridF, 256, 0, stream>>>(
                as_hi, as_lo, bs_hi, bs_lo, b1 + (size_t)t * HFF, ffnb, HN, HFF, HH);
            gelu_kernel<<<cdiv((long long)HN * HFF, 256), 256, 0, stream>>>(
                ffnb, (long long)HN * HFF);

            // h2 = f @ W2 + b2
            split_bf16_kernel<<<cdiv((long long)HN * HFF, 256), 256, 0, stream>>>(
                ffnb, as_hi, as_lo, (long long)HN * HFF);
            split_bf16_t_kernel<<<cdiv(szHF, 256), 256, 0, stream>>>(
                W2 + t * szHF, bs_hi, bs_lo, HFF, HH);
            gemm_bf16x3_kernel<<<gridH, 256, 0, stream>>>(
                as_hi, as_lo, bs_hi, bs_lo, b2 + (size_t)t * HH, h2_t, HN, HH, HFF);

            // out = LN2(x1 + h2)
            add_ln_kernel<<<lnBlocks, 256, 0, stream>>>(
                x1_t, h2_t, ln2_g + (size_t)t * HH, ln2_b + (size_t)t * HH,
                out + (size_t)t * NH, HN);
        }
    }
}
